// GeoEEGSNN_24910810317392
// MI455X (gfx1250) — matrix operand order repaired and checked by reading, not compiled, not run on hardware
//
#include <hip/hip_runtime.h>
#include <hip/hip_bf16.h>

typedef __attribute__((ext_vector_type(16))) __bf16 v16bf;
typedef __attribute__((ext_vector_type(8)))  float  v8f;
typedef __attribute__((ext_vector_type(8)))  unsigned short u16x8;
typedef __attribute__((ext_vector_type(4)))  unsigned short u16x4;
typedef __attribute__((ext_vector_type(4)))  float  float4v;

// ---------------------------------------------------------------------------
#define TT    2048   // time steps
#define BB    32     // batch
#define CC    64     // channels
#define TPOOL 256    // T/8

__device__ __forceinline__ unsigned short f2bf(float f) {
    __bf16 h = (__bf16)f;
    return *(unsigned short*)&h;
}

// ---------------------------------------------------------------------------
// Weight fp32 -> bf16 cast.  Packed layout in one bf16 buffer:
//   [0,      98304)  sconv1_w (128x768)
//   [98304, 106496)  sconv2_w (64x128)
//   [106496,110592)  tconv2_w (64x64)
//   [110592,172032)  tconv1_w (64x(64*15))  == im2col'd W' directly
// ---------------------------------------------------------------------------
__global__ void convert_weights_kernel(const float* __restrict__ w1,
                                       const float* __restrict__ w2,
                                       const float* __restrict__ w4,
                                       const float* __restrict__ w3,
                                       unsigned short* __restrict__ out) {
    int idx = blockIdx.x * 256 + threadIdx.x;
    float v;
    if      (idx < 98304)  v = w1[idx];
    else if (idx < 106496) v = w2[idx - 98304];
    else if (idx < 110592) v = w4[idx - 106496];
    else if (idx < 172032) v = w3[idx - 110592];
    else return;
    out[idx] = f2bf(v);
}

// ---------------------------------------------------------------------------
// Population encoding: xb (B,3,C,T) -> x_enc bf16 (B, 3*C*4, T)
// channel = f*(C*4) + c*4 + g.  Each thread: 4 consecutive t x 4 gaussians.
// ---------------------------------------------------------------------------
__global__ __launch_bounds__(256) void encode_kernel(const float* __restrict__ xb,
                                                     const float* __restrict__ mu,
                                                     const float* __restrict__ sigma,
                                                     unsigned short* __restrict__ xe) {
    size_t idx = (size_t)blockIdx.x * 256 + threadIdx.x;   // over B*3*C*(T/4)
    int t4 = (int)(idx & (TT / 4 - 1));                    // 9 bits
    size_t rest = idx >> 9;
    int c = (int)(rest & (CC - 1));
    rest >>= 6;
    int f = (int)(rest % 3);
    int b = (int)(rest / 3);

    const float* src = xb + ((((size_t)b * 3 + f) * CC + c) * TT) + t4 * 4;
    float4v x = *(const float4v*)src;

#pragma unroll
    for (int g = 0; g < 4; ++g) {
        float den = 2.0f * sigma[g] * sigma[g] + 1e-6f;
        float mg  = mu[g];
        u16x4 o;
#pragma unroll
        for (int j = 0; j < 4; ++j) {
            float d = x[j] - mg;
            o[j] = f2bf(__expf(-d * d / den));
        }
        *(u16x4*)&xe[((size_t)b * 768 + f * 256 + c * 4 + g) * TT + t4 * 4] = o;
    }
}

// ---------------------------------------------------------------------------
// Pooled curvature/tangent streams: (Tp, B, C) layout
// ---------------------------------------------------------------------------
__global__ __launch_bounds__(256) void pool_streams_kernel(const float* __restrict__ xb,
                                                           float* __restrict__ curv,
                                                           float* __restrict__ tang) {
    int idx = blockIdx.x * 256 + threadIdx.x;     // over Tp*B*C = 524288
    int c  = idx & 63;
    int b  = (idx >> 6) & 31;
    int tp = idx >> 11;
    const float* pc = xb + (((size_t)b * 3 + 1) * CC + c) * TT + tp * 8;
    const float* pt = xb + (((size_t)b * 3 + 2) * CC + c) * TT + tp * 8;
    float s1 = 0.f, s2 = 0.f;
#pragma unroll
    for (int j = 0; j < 8; ++j) { s1 += pc[j]; s2 += pt[j]; }
    curv[idx] = s1 * 0.125f;   // idx == (tp*B + b)*C + c
    tang[idx] = s2 * 0.125f;
}

// ---------------------------------------------------------------------------
// Generic bf16 WMMA GEMM:  Y[b] (MxT, f32) = W (MxK, bf16) * X[b] (KxT, bf16)
//   kw==1 : plain 1x1 conv GEMM, X is (B, K, T)
//   kw==15: im2col temporal conv (pad 7), X is (B, 64, T), K = 64*15 = 960
// Block: 256 threads (8 waves). Tile: 64(M) x 128(N). Each wave: 16-wide N
// strip, 4 M-subtiles of 16x16, K stepped by 32.
// W tile is staged with CDNA5 async memory->LDS copies (ASYNCcnt); X tile is
// fetched with b128 vector loads and transposed into LDS.
// ---------------------------------------------------------------------------
__global__ __launch_bounds__(256) void gemm_bf16_wmma(const unsigned short* __restrict__ W,
                                                      const unsigned short* __restrict__ X,
                                                      float* __restrict__ Y,
                                                      int M, int K, int kw) {
    __shared__ unsigned short Wlds[64 * 34];    // [m][k], row stride 34 elems (68 B)
    __shared__ unsigned short Xlds[128 * 34];   // [n][k] (transposed), stride 34

    const int tid  = threadIdx.x;
    const int lane = tid & 31;
    const int wave = tid >> 5;
    const int half = lane >> 4;
    const int l16  = lane & 15;

    const int bz    = blockIdx.z;
    const int mbase = blockIdx.y * 64;
    const int t0    = blockIdx.x * 128;
    const int noff  = wave * 16;

    const unsigned wlds_base = (unsigned)(size_t)&Wlds[0];

    v8f acc[4] = {};

    for (int k0 = 0; k0 < K; k0 += 32) {
        // ---- stage W tile 64x32 via async global->LDS (1024 dwords) ----
#pragma unroll
        for (int i = 0; i < 4; ++i) {
            int d   = tid + i * 256;           // dword index
            int row = d >> 4;                  // 16 dwords (32 bf16) per row
            int col = d & 15;                  // dword column
            unsigned long long ga =
                (unsigned long long)(size_t)(W + (size_t)(mbase + row) * K + k0) +
                (unsigned)(col * 4);
            unsigned la = wlds_base + (unsigned)(row * 68 + col * 4);
            asm volatile("global_load_async_to_lds_b32 %0, %1, off"
                         :: "v"(la), "v"(ga) : "memory");
        }

        // ---- stage X tile 32(K) x 128(N), transposed into [n][k] ----
        if (kw == 1) {
#pragma unroll
            for (int i = 0; i < 2; ++i) {
                int s   = tid + i * 256;       // 512 vector slots of 8 elems
                int kl  = s >> 4;              // 0..31
                int tl0 = (s & 15) * 8;        // 0..120
                u16x8 v = *(const u16x8*)&X[((size_t)bz * K + (k0 + kl)) * TT +
                                            (t0 + tl0)];
#pragma unroll
                for (int j = 0; j < 8; ++j) Xlds[(tl0 + j) * 34 + kl] = v[j];
            }
        } else {  // im2col for k=15 temporal conv, pad 7 (boundary-safe scalar)
#pragma unroll
            for (int i = 0; i < 16; ++i) {
                int e  = tid + i * 256;
                int tl = e & 127, kl = e >> 7;
                int kk = k0 + kl;
                int c = kk / 15, r = kk - c * 15;
                int tg = t0 + tl + r - 7;
                unsigned short v = (tg >= 0 && tg < TT)
                    ? X[((size_t)bz * 64 + c) * TT + tg] : (unsigned short)0;
                Xlds[tl * 34 + kl] = v;
            }
        }
        asm volatile("s_wait_asynccnt 0x0" ::: "memory");
        __syncthreads();

        // ---- B fragment (32x16) for this wave's N strip ----
        union { unsigned int u[8]; v16bf v; } bf;
#pragma unroll
        for (int v = 0; v < 8; ++v) {
            int k = (v < 4) ? (2 * v + 8 * half) : (2 * v + 8 + 8 * half);
            bf.u[v] = *(const unsigned int*)&Xlds[(noff + l16) * 34 + k];
        }
        // ---- A fragments (16x32) for 4 M-subtiles + WMMA ----
#pragma unroll
        for (int s = 0; s < 4; ++s) {
            union { unsigned int u[8]; v16bf v; } af;
#pragma unroll
            for (int v = 0; v < 8; ++v) {
                int k = (v < 4) ? (2 * v + 8 * half) : (2 * v + 8 + 8 * half);
                af.u[v] = *(const unsigned int*)&Wlds[(s * 16 + l16) * 34 + k];
            }
            acc[s] = __builtin_amdgcn_wmma_f32_16x16x32_bf16(
                false, af.v, false, bf.v, (short)0, acc[s], false, false);
        }
        __syncthreads();
    }

    // ---- epilogue: D layout (f32 16x16): m = v + 8*half, n = lane&15 ----
    float* Yb = Y + (size_t)bz * M * TT;
#pragma unroll
    for (int s = 0; s < 4; ++s) {
#pragma unroll
        for (int v = 0; v < 8; ++v) {
            int m = mbase + s * 16 + v + 8 * half;
            Yb[(size_t)m * TT + (t0 + noff + l16)] = acc[s][v];
        }
    }
}

// ---------------------------------------------------------------------------
// Fused InstanceNorm (biased var, eps 1e-5) + ReLU.
//   Pool == nullptr : write bf16 activations (input for next GEMM)
//   Pool != nullptr : also AvgPool1d(8) and write (Tp,B,C) f32 instead
// One block per (b, channel); 256 threads x 8 consecutive t each.
// ---------------------------------------------------------------------------
__global__ __launch_bounds__(256) void in_relu_kernel(const float* __restrict__ Xf,
                                                      const float* __restrict__ gw,
                                                      const float* __restrict__ gb,
                                                      unsigned short* __restrict__ Yb,
                                                      float* __restrict__ Pool,
                                                      int M) {
    __shared__ float red[256];
    __shared__ float red2[256];
    int bm = blockIdx.x;
    int b = bm / M, m = bm % M;
    const float* x = Xf + ((size_t)b * M + m) * TT;
    int tid = threadIdx.x;

    float v[8];
    float s = 0.f, ss = 0.f;
#pragma unroll
    for (int j = 0; j < 8; ++j) {
        v[j] = x[tid * 8 + j];
        s += v[j];
        ss += v[j] * v[j];
    }
    red[tid] = s; red2[tid] = ss;
    __syncthreads();
    for (int st = 128; st > 0; st >>= 1) {
        if (tid < st) { red[tid] += red[tid + st]; red2[tid] += red2[tid + st]; }
        __syncthreads();
    }
    float mean = red[0] * (1.0f / TT);
    float var  = red2[0] * (1.0f / TT) - mean * mean;
    float inv  = rsqrtf(var + 1e-5f) * gw[m];
    float bias = gb[m];

    if (Pool) {
        float acc = 0.f;
#pragma unroll
        for (int j = 0; j < 8; ++j) {
            float y = (v[j] - mean) * inv + bias;
            acc += (y > 0.f ? y : 0.f);
        }
        Pool[((size_t)tid * BB + b) * M + m] = acc * 0.125f;  // (Tp,B,C)
    } else {
        unsigned short* y = Yb + ((size_t)b * M + m) * TT;
        u16x8 o;
#pragma unroll
        for (int j = 0; j < 8; ++j) {
            float t = (v[j] - mean) * inv + bias;
            o[j] = f2bf(t > 0.f ? t : 0.f);
        }
        *(u16x8*)&y[tid * 8] = o;
    }
}

// ---------------------------------------------------------------------------
// GD-LIF scan over T'=256 steps + softmax attention, parallel over (b,c).
// out_feat[b,c] = sum_t spike_t * exp(curv_t - max) / sum_t exp(curv_t - max)
// ---------------------------------------------------------------------------
__global__ __launch_bounds__(256) void scan_kernel(const float* __restrict__ cin,
                                                   const float* __restrict__ curv,
                                                   const float* __restrict__ tang,
                                                   const float* __restrict__ alpha_p,
                                                   const float* __restrict__ gamma_p,
                                                   const float* __restrict__ bl_p,
                                                   float* __restrict__ out_feat) {
    const int BC = BB * CC;
    int idx = blockIdx.x * 256 + threadIdx.x;
    if (idx >= BC) return;
    float alpha = alpha_p[0], gamma = gamma_p[0], bl = bl_p[0];

    float mx = -3.4e38f;
    for (int t = 0; t < TPOOL; ++t) {
        float cv = curv[t * BC + idx];
        mx = cv > mx ? cv : mx;
    }
    float m = 0.f, num = 0.f, den = 0.f;
    for (int t = 0; t < TPOOL; ++t) {
        float cv = curv[t * BC + idx];
        float tg = tang[t * BC + idx];
        float xt = cin[t * BC + idx];
        float vth  = 1.0f + alpha * tg;
        float beta = 1.0f / (1.0f + __expf(-(bl - gamma * cv)));
        m = beta * m + xt;
        float spike = (m - vth > 0.f) ? 1.f : 0.f;
        m -= spike * vth;
        float e = __expf(cv - mx);
        den += e;
        num += spike * e;
    }
    out_feat[idx] = num / den;
}

// ---------------------------------------------------------------------------
// Final FC: (B,64) @ (4,64)^T + bias -> (B,4)
// ---------------------------------------------------------------------------
__global__ void fc_kernel(const float* __restrict__ feat,
                          const float* __restrict__ fw,
                          const float* __restrict__ fb,
                          float* __restrict__ out) {
    int idx = threadIdx.x;
    if (idx >= 128) return;
    int b = idx >> 2, j = idx & 3;
    float s = fb[j];
#pragma unroll
    for (int c = 0; c < 64; ++c) s += feat[b * 64 + c] * fw[j * 64 + c];
    out[idx] = s;
}

// ---------------------------------------------------------------------------
extern "C" void kernel_launch(void* const* d_in, const int* in_sizes, int n_in,
                              void* d_out, int out_size, void* d_ws, size_t ws_size,
                              hipStream_t stream) {
    const float* xb         = (const float*)d_in[0];
    const float* mu         = (const float*)d_in[1];
    const float* sigma      = (const float*)d_in[2];
    const float* sconv1_w   = (const float*)d_in[3];
    const float* in1_w      = (const float*)d_in[4];
    const float* in1_b      = (const float*)d_in[5];
    const float* sconv2_w   = (const float*)d_in[6];
    const float* in2_w      = (const float*)d_in[7];
    const float* in2_b      = (const float*)d_in[8];
    const float* tconv1_w   = (const float*)d_in[9];
    const float* in3_w      = (const float*)d_in[10];
    const float* in3_b      = (const float*)d_in[11];
    const float* tconv2_w   = (const float*)d_in[12];
    const float* in4_w      = (const float*)d_in[13];
    const float* in4_b      = (const float*)d_in[14];
    const float* alpha      = (const float*)d_in[15];
    const float* gamma      = (const float*)d_in[16];
    const float* beta_logit = (const float*)d_in[17];
    const float* fc_w       = (const float*)d_in[18];
    const float* fc_b       = (const float*)d_in[19];
    float* out = (float*)d_out;

    char* ws = (char*)d_ws;
    // Region [0, 100663296): x_enc during stage 1, recycled afterwards.
    unsigned short* x_enc = (unsigned short*)(ws + 0);        // 100,663,296 B
    float*          h2    = (float*)         (ws + 0);        // 16.8 MB (x_enc dead)
    unsigned short* h2b   = (unsigned short*)(ws + 16777216); //  8.4 MB
    float*          h3    = (float*)         (ws + 25165824); // 16.8 MB
    unsigned short* h3b   = (unsigned short*)(ws + 41943040); //  8.4 MB
    float*          h4    = (float*)         (ws + 50331648); // 16.8 MB
    // Persistent region beyond x_enc:
    unsigned short* wb    = (unsigned short*)(ws + 100663296);// 344,064 B (padded)
    float*          h1    = (float*)         (ws + 101187584);// 33.6 MB
    unsigned short* h1b   = (unsigned short*)(ws + 134742016);// 16.8 MB
    float*          cpool = (float*)         (ws + 151519232);//  2.1 MB (Tp,B,C)
    float*          curvp = (float*)         (ws + 153616384);//  2.1 MB
    float*          tangp = (float*)         (ws + 155713536);//  2.1 MB
    float*          feat  = (float*)         (ws + 157810688);//  8 KB

    // 1. prep
    convert_weights_kernel<<<672, 256, 0, stream>>>(sconv1_w, sconv2_w,
                                                    tconv2_w, tconv1_w, wb);
    encode_kernel<<<12288, 256, 0, stream>>>(xb, mu, sigma, x_enc);
    pool_streams_kernel<<<2048, 256, 0, stream>>>(xb, curvp, tangp);

    // 2. sconv1: (128 x 768) GEMM + IN1 + relu
    dim3 g1(16, 2, 32);
    gemm_bf16_wmma<<<g1, 256, 0, stream>>>(wb, x_enc, h1, 128, 768, 1);
    in_relu_kernel<<<32 * 128, 256, 0, stream>>>(h1, in1_w, in1_b, h1b, nullptr, 128);

    // 3. sconv2: (64 x 128) GEMM + IN2 + relu
    dim3 g2(16, 1, 32);
    gemm_bf16_wmma<<<g2, 256, 0, stream>>>(wb + 98304, h1b, h2, 64, 128, 1);
    in_relu_kernel<<<32 * 64, 256, 0, stream>>>(h2, in2_w, in2_b, h2b, nullptr, 64);

    // 4. tconv1 k=15 pad=7 as im2col GEMM (64 x 960) + IN3 + relu
    gemm_bf16_wmma<<<g2, 256, 0, stream>>>(wb + 110592, h2b, h3, 64, 960, 15);
    in_relu_kernel<<<32 * 64, 256, 0, stream>>>(h3, in3_w, in3_b, h3b, nullptr, 64);

    // 5. tconv2: (64 x 64) GEMM + IN4 + relu + avgpool8 -> (Tp,B,C)
    gemm_bf16_wmma<<<g2, 256, 0, stream>>>(wb + 106496, h3b, h4, 64, 64, 1);
    in_relu_kernel<<<32 * 64, 256, 0, stream>>>(h4, in4_w, in4_b, nullptr, cpool, 64);

    // 6. LIF scan + softmax attention + FC
    scan_kernel<<<8, 256, 0, stream>>>(cpool, curvp, tangp, alpha, gamma,
                                       beta_logit, feat);
    fc_kernel<<<1, 128, 0, stream>>>(feat, fc_w, fc_b, out);
}
